// MVPROPFAT3D_22797686408172
// MI455X (gfx1250) — matrix operand order repaired and checked by reading, not compiled, not run on hardware
//
#include <hip/hip_runtime.h>
#include <hip/hip_bf16.h>
#include <math.h>

typedef __attribute__((ext_vector_type(16))) _Float16 v16h;
typedef __attribute__((ext_vector_type(8)))  float    v8f;

#define NB 4
#define ND 64
#define NH 64
#define NW 64
#define NCH 150
#define KITERS 30

__device__ __forceinline__ int clamp63(int x) {
    x = x < 0 ? 0 : x;
    return x > 63 ? 63 : x;
}

// ---------------------------------------------------------------------------
// Phase 1: fused conv3x3x3 (1->150 ch) + bias + relu + channel-weighted sum +
// sigmoid -> p[voxel]; also initializes v0 = r.
// One wave = one 16-voxel tile along W. GEMM via v_wmma_f32_16x16x32_f16:
//   A[16 voxels x K=32]  (27 im2col taps, zero padded)  f16
//   B[K=32 x 16 channels] (weights, zero padded)        f16, staged in LDS
// 10 channel-tiles cover 150 channels. Bias/gate tables are zero-padded in
// LDS so the epilogue is fully branch-free.
// ---------------------------------------------------------------------------
__global__ void __launch_bounds__(256)
conv_gate_kernel(const float* __restrict__ image,   // (B,2,D,H,W)
                 const float* __restrict__ hw,      // (150,27)
                 const float* __restrict__ hb,      // (150)
                 const float* __restrict__ pw,      // (150)
                 float* __restrict__ p_out,         // (B,D,H,W)
                 float* __restrict__ v_out)         // (B,D,H,W)
{
    // LDS-staged B tiles: [ct][lane][i], per-lane v16h loadable as 2x b128.
    __shared__ __attribute__((aligned(32))) _Float16 sB[10][32][16];
    __shared__ float sPW[160];   // zero-padded gate weights
    __shared__ float sHB[160];   // zero-padded biases

    const int tid = threadIdx.x;

    // ---- Cooperative prologue (once per block) ----------------------------
    for (int e = tid; e < 10 * 32 * 16; e += 256) {   // 5120 halfs of B
        const int ct  = e >> 9;             // / 512
        const int rem = e & 511;
        const int ln  = rem >> 4;           // lane slot
        const int i   = rem & 15;
        const int k   = ((ln < 16) ? 0 : 8) + ((i < 8) ? i : (i + 8));
        const int ch  = ct * 16 + (ln & 15);
        float val = 0.0f;
        if (k < 27 && ch < NCH) val = hw[ch * 27 + k];
        sB[ct][ln][i] = (_Float16)val;
    }
    if (tid < 160) {
        const bool v = tid < NCH;
        sPW[tid] = v ? pw[tid] : 0.0f;
        sHB[tid] = v ? hb[tid] : 0.0f;
    }
    __syncthreads();

    const int lane = tid & 31;
    const int wave = tid >> 5;
    const int tile = blockIdx.x * 8 + wave;     // 65536 tiles total

    const int w0 = (tile & 3) * 16;             // W/16 = 4
    const int h  = (tile >> 2) & 63;
    const int d  = (tile >> 8) & 63;
    const int b  = (tile >> 14);

    const int m  = lane & 15;                   // row of A (voxel in tile)
    const int w  = w0 + m;

    const size_t plane    = (size_t)NH * NW;
    const size_t vox_base = (((size_t)b * ND + d) * NH + h) * NW;
    const size_t x_base   = (((size_t)b * 2 + 0) * ND) * plane;
    const size_t r_base   = (((size_t)b * 2 + 1) * ND) * plane;

    // v0 = r  (lanes 0..15 cover the 16 voxels of this tile)
    if (lane < 16) {
        v_out[vox_base + w] = image[r_base + d * plane + (size_t)h * NW + w];
    }

    // ---- Build A (im2col patch), branch-free ------------------------------
    // lanes 0-15:  halfs 0..7 -> K=0..7,   halfs 8..15 -> K=16..23
    // lanes 16-31: halfs 0..7 -> K=8..15,  halfs 8..15 -> K=24..31
    const int kb = (lane < 16) ? 0 : 8;
    v16h a;
#pragma unroll
    for (int i = 0; i < 16; ++i) {
        const int k  = (i < 8) ? (kb + i) : (kb + 16 + (i - 8));
        const int kk = k < 27 ? k : 0;          // clamp tap id
        const int dz = kk / 9;
        const int dy = (kk / 3) % 3;
        const int dx = kk % 3;
        const int dd = d + dz - 1;
        const int hh = h + dy - 1;
        const int ww = w + dx - 1;
        const bool ok = (k < 27) &
                        ((unsigned)dd < 64u) & ((unsigned)hh < 64u) &
                        ((unsigned)ww < 64u);
        // Always load from a clamped (valid) address; select 0 if OOB/pad.
        const float val = image[x_base + (size_t)clamp63(dd) * plane +
                                (size_t)clamp63(hh) * NW + clamp63(ww)];
        a[i] = (_Float16)(ok ? val : 0.0f);
    }

    float s[8];
#pragma unroll
    for (int j = 0; j < 8; ++j) s[j] = 0.0f;

    const int n = lane & 15;                    // column of B (channel in tile)

#pragma unroll
    for (int ct = 0; ct < 10; ++ct) {
        const int ch = ct * 16 + n;

        const v16h bm = *(const v16h*)&sB[ct][lane][0];   // 2x ds_load_b128

        v8f c = {};
        c = __builtin_amdgcn_wmma_f32_16x16x32_f16(
                /*neg_a=*/false, a, /*neg_b=*/false, bm,
                /*c_mod=*/(short)0, c, /*reuse_a=*/false, /*reuse_b=*/false);

        // Per-channel epilogue: relu(acc + bias) * pw, accumulated over N.
        // C layout: VGPR j, lanes 0-15 -> (M=j, N=lane); lanes 16-31 -> M=8+j.
        // Tables are zero-padded (ch in [150,160)) -> no guards needed.
        const float bias = sHB[ch];
        const float wgt  = sPW[ch];
#pragma unroll
        for (int j = 0; j < 8; ++j) {
            float hv = c[j] + bias;
            hv = hv > 0.0f ? hv : 0.0f;
            s[j] = fmaf(wgt, hv, s[j]);
        }
    }

    // Reduce over N (channels) = over the 16 lanes of each half-wave.
#pragma unroll
    for (int off = 1; off < 16; off <<= 1) {
#pragma unroll
        for (int j = 0; j < 8; ++j) s[j] += __shfl_xor(s[j], off, 32);
    }

    if (n == 0) {
        const int mrow0 = (lane < 16) ? 0 : 8;  // lane 0 -> M=0..7, lane 16 -> 8..15
#pragma unroll
        for (int j = 0; j < 8; ++j) {
            const float sig = 1.0f / (1.0f + __expf(-s[j]));
            p_out[vox_base + w0 + mrow0 + j] = sig;
        }
    }
}

// ---------------------------------------------------------------------------
// Phase 2: one propagation step
//   v_new = max(v_old, r + p * (maxpool3(v_old) - r))
// Replicate-clamp padding == -inf padding for a max window (the clamped tap
// duplicates an in-window sample), so the stencil is fully branch-free:
// 27 unconditional loads + a v_max chain. v and p (4 MB each) are L2-resident.
// ---------------------------------------------------------------------------
__global__ void __launch_bounds__(256)
prop_kernel(const float* __restrict__ v_in,
            const float* __restrict__ image,
            const float* __restrict__ p,
            float* __restrict__ v_out)
{
    const int w  = threadIdx.x;                         // 0..63
    const int h  = blockIdx.y * blockDim.y + threadIdx.y;
    const int bd = blockIdx.z;                          // b*64 + d
    const int d  = bd & 63;
    const int b  = bd >> 6;

    const size_t plane = (size_t)NH * NW;
    const size_t idx = ((size_t)bd * NH + h) * NW + w;

    const int wm = clamp63(w - 1), wp = clamp63(w + 1);

    const float vc = v_in[idx];
    float mx = vc;
#pragma unroll
    for (int dz = -1; dz <= 1; ++dz) {
        const int dd = clamp63(d + dz);
#pragma unroll
        for (int dy = -1; dy <= 1; ++dy) {
            const int hh = clamp63(h + dy);
            const float* row = v_in + (((size_t)b * ND + dd) * NH + hh) * NW;
            const float a0 = row[wm];
            const float a1 = row[w];
            const float a2 = row[wp];
            mx = fmaxf(mx, fmaxf(a0, fmaxf(a1, a2)));
        }
    }

    const float r = image[(((size_t)b * 2 + 1) * ND + d) * plane +
                          (size_t)h * NW + w];
    const float cand = fmaf(p[idx], mx - r, r);
    v_out[idx] = vc > cand ? vc : cand;
}

// ---------------------------------------------------------------------------
extern "C" void kernel_launch(void* const* d_in, const int* in_sizes, int n_in,
                              void* d_out, int out_size, void* d_ws, size_t ws_size,
                              hipStream_t stream)
{
    const float* image = (const float*)d_in[0];   // (4,2,64,64,64)
    const float* hw    = (const float*)d_in[1];   // (150,1,3,3,3)
    const float* hb    = (const float*)d_in[2];   // (150,)
    const float* pw    = (const float*)d_in[3];   // (1,150,1,1,1)
    // d_in[4] = k (device scalar); fixed at 30 per the reference — cannot be
    // read host-side during graph capture.

    float* v_final = (float*)d_out;               // (4,64,64,64)
    float* p_buf   = (float*)d_ws;                              // 4 MB
    float* v_alt   = (float*)d_ws + (size_t)NB * ND * NH * NW;  // 4 MB

    // Phase 1: p + v0 (65536 wave-tiles, 8 waves/block)
    conv_gate_kernel<<<8192, 256, 0, stream>>>(image, hw, hb, pw, p_buf, v_final);

    // Phase 2: 30 ping-pong stencil steps; even count -> result ends in d_out.
    dim3 blk(64, 4, 1);
    dim3 grd(1, NH / 4, NB * ND);
    float* va = v_final;
    float* vb = v_alt;
    for (int it = 0; it < KITERS; ++it) {
        prop_kernel<<<grd, blk, 0, stream>>>(va, image, p_buf, vb);
        float* t = va; va = vb; vb = t;
    }
}
